// TVKoopmanMoE_85538568667812
// MI455X (gfx1250) — matrix operand order repaired and checked by reading, not compiled, not run on hardware
//
#include <hip/hip_runtime.h>
#include <hip/hip_bf16.h>

// ---------------- problem constants ----------------
#define BB     4096
#define CC     128
#define NSEG_  3
#define NPAD   10288        // 8192(S)+2048(L)+8(G)+32(Sig)=10280 padded to 16*643
#define NTILES 643
#define BCH    512          // batch chunk
#define NCHUNK 8
#define MU_    0.14861456f  // -ln(0.7)/(0.1*24)
#define SQ_SCALE (0.1f/64.0f) // DT / 2^SQ

typedef __attribute__((ext_vector_type(2)))  float    v2f;
typedef __attribute__((ext_vector_type(4)))  float    v4f;
typedef __attribute__((ext_vector_type(8)))  float    v8f;
typedef __attribute__((ext_vector_type(8)))  _Float16 v8h;
typedef __attribute__((ext_vector_type(16))) _Float16 v16h;

// ---------------- f32 -> f16 context convert ----------------
__global__ void k_cvt_ctx(const float* __restrict__ ctx, _Float16* __restrict__ ctx16) {
  int i = blockIdx.x * 256 + threadIdx.x;
  if (i < BB * CC) ctx16[i] = (_Float16)ctx[i];
}

// ---------------- build transposed f16 weight panel W16T[n][c] ----------------
__global__ void k_build_w16t(const float* __restrict__ SW, const float* __restrict__ LW,
                             const float* __restrict__ GW, const float* __restrict__ SigW,
                             _Float16* __restrict__ W16T, int seg) {
  int idx = blockIdx.x * 256 + threadIdx.x;
  if (idx >= NPAD * CC) return;
  int n = idx >> 7;
  int c = idx & 127;
  size_t base = (size_t)seg * CC + c;
  float v = 0.0f;
  if (n < 8192)        v = SW[base * 8192 + n];
  else if (n < 10240)  v = LW[base * 2048 + (n - 8192)];
  else if (n < 10248)  v = GW[base * 8 + (n - 10240)];
  else if (n < 10280)  v = SigW[base * 32 + (n - 10248)];
  W16T[idx] = (_Float16)v;
}

// ---------------- big context GEMM: Y = ctx16 @ W16  (f16 WMMA, f32 acc) ----------------
__global__ void __launch_bounds__(128) k_gemm(const _Float16* __restrict__ ctx16,
                                              const _Float16* __restrict__ w16t,
                                              float* __restrict__ Y, int b0) {
  int wave = threadIdx.x >> 5;
  int lane = threadIdx.x & 31;
  int nTile = blockIdx.y * 4 + wave;
  if (nTile >= NTILES) return;               // wave-uniform, EXEC stays full
  int hi  = lane >> 4;
  int l15 = lane & 15;
  int rowTile = blockIdx.x;
  const _Float16* arow = ctx16 + (size_t)(b0 + rowTile * 16 + l15) * CC;
  const _Float16* brow = w16t  + (size_t)(nTile * 16 + l15) * CC;
  v8f acc = {};
#pragma unroll
  for (int kc = 0; kc < 4; ++kc) {
    int kb = kc * 32;
    // 16-bit A 16x32: lanes<16 -> K 0..15 ; lanes>=16 -> K 16..31 (contiguous, matches B)
    v16h af = *(const v16h*)(arow + kb + hi * 16);
    // 16-bit B 32x16: lanes<16 -> K 0..15 ; lanes>=16 -> K 16..31 (contiguous)
    v16h bf = *(const v16h*)(brow + kb + hi * 16);
    acc = __builtin_amdgcn_wmma_f32_16x16x32_f16(false, af, false, bf, (short)0, acc, false, false);
  }
#pragma unroll
  for (int r = 0; r < 8; ++r)
    Y[(size_t)(rowTile * 16 + r + hi * 8) * NPAD + nTile * 16 + l15] = acc[r];
}

// ---------------- assemble A (skew - LL^T - mu I, gated, frob-clamped) + qdiag ----------------
__global__ void __launch_bounds__(128) k_assemble(const float* __restrict__ Y,
    const float* __restrict__ Sb, const float* __restrict__ Lb,
    const float* __restrict__ Gb, const float* __restrict__ Sigb,
    float* __restrict__ Aws, float* __restrict__ qdws, int seg) {
  __shared__ float sL[2048];
  __shared__ float sA[1024];
  __shared__ float sw[8];
  __shared__ float sqd[32];
  __shared__ float rbuf[128];
  int b = blockIdx.x;
  int t = threadIdx.x;
  const float* y = Y + (size_t)b * NPAD;
  for (int i = t; i < 2048; i += 128) sL[i] = y[8192 + i] + Lb[seg * 2048 + i];
  if (t == 0) {
    float l[8]; float mx = -1e30f;
#pragma unroll
    for (int k = 0; k < 8; ++k) { l[k] = (y[10240 + k] + Gb[seg * 8 + k]) * 0.5f; mx = fmaxf(mx, l[k]); }
    float ss = 0.0f;
#pragma unroll
    for (int k = 0; k < 8; ++k) { l[k] = expf(l[k] - mx); ss += l[k]; }
#pragma unroll
    for (int k = 0; k < 8; ++k) sw[k] = l[k] / ss;
  }
  if (t < 32) {
    float x = y[10248 + t] + Sigb[seg * 32 + t];
    float sp = (x > 20.0f) ? x : log1pf(expf(x));
    float dg = fminf(sp, 1.0f);
    sqd[t] = dg * dg;
  }
  __syncthreads();
  float w[8];
#pragma unroll
  for (int k = 0; k < 8; ++k) w[k] = sw[k];
  float ps = 0.0f;
  for (int p = t; p < 1024; p += 128) {
    int d = p >> 5, e = p & 31;
    float a = (d == e) ? -MU_ : 0.0f;
#pragma unroll
    for (int k = 0; k < 8; ++k) {
      float sde = y[k * 1024 + d * 32 + e] + Sb[seg * 8192 + k * 1024 + d * 32 + e];
      float sed = y[k * 1024 + e * 32 + d] + Sb[seg * 8192 + k * 1024 + e * 32 + d];
      const float* Ld = sL + k * 256 + d * 8;
      const float* Le = sL + k * 256 + e * 8;
      float llt = 0.0f;
#pragma unroll
      for (int r = 0; r < 8; ++r) llt += Ld[r] * Le[r];
      a += w[k] * (sde - sed - llt);
    }
    sA[p] = a;
    ps += a * a;
  }
  rbuf[t] = ps;
  __syncthreads();
  for (int off = 64; off > 0; off >>= 1) {
    if (t < off) rbuf[t] += rbuf[t + off];
    __syncthreads();
  }
  float frob = sqrtf(rbuf[0]);
  float scale = (frob > 3.0f) ? (3.0f / (frob + 1e-6f)) : 1.0f;
  for (int p = t; p < 1024; p += 128) Aws[(size_t)b * 1024 + p] = sA[p] * scale;
  if (t < 32) qdws[b * 32 + t] = sqd[t];
}

// =============================================================================
// per-wave 32x32 f32 matmul machinery, column-major LDS slabs: addr(r,c)=c*32+r
// =============================================================================
struct PF { v2f f[16]; };   // A-operand fragments, index [i*8+kc]
struct BF { v2f f[16]; };   // B-operand fragments, index [j*8+kc]

__device__ __forceinline__ void loadPF(const float* S, int o, int lane, PF& pf) {
  const int hi = lane >> 4, l15 = lane & 15;
#pragma unroll
  for (int i = 0; i < 2; ++i)
#pragma unroll
    for (int kc = 0; kc < 8; ++kc) {
      int k0 = kc * 4 + hi * 2;                 // f32 A 16x4: VGPR0=K(2hi), VGPR1=K(2hi+1)
      v2f a;
      a.x = S[o + k0 * 32 + i * 16 + l15];
      a.y = S[o + (k0 + 1) * 32 + i * 16 + l15];
      pf.f[i * 8 + kc] = a;
    }
}
__device__ __forceinline__ void loadBF(const float* S, int o, int lane, BF& bf) {
  const int hi = lane >> 4, l15 = lane & 15;
#pragma unroll
  for (int j = 0; j < 2; ++j)
#pragma unroll
    for (int kc = 0; kc < 8; ++kc) {
      int k0 = kc * 4 + hi * 2;                 // f32 B 4x16: rows 2hi, 2hi+1 (contiguous in col-major)
      bf.f[j * 8 + kc] = *(const v2f*)&S[o + (j * 16 + l15) * 32 + k0];
    }
}

// CMODE: 0 = no C, 1 = C from LDS (cOff), 2 = C from regs (creg[4])
template<int CMODE>
__device__ __forceinline__ void mm32x(float* S, const PF& pf, const BF& bf,
                                      const v8f* creg, int cOff, int yOff, int lane) {
  const int hi = lane >> 4, l15 = lane & 15;
#pragma unroll
  for (int i = 0; i < 2; ++i) {
#pragma unroll
    for (int j = 0; j < 2; ++j) {
      int cb = (j * 16 + l15) * 32 + i * 16 + hi * 8;   // contiguous r=0..7
      v8f acc;
      if constexpr (CMODE == 1) {
        v4f c0 = *(const v4f*)&S[cOff + cb];
        v4f c1 = *(const v4f*)&S[cOff + cb + 4];
        acc = __builtin_shufflevector(c0, c1, 0,1,2,3,4,5,6,7);
      } else if constexpr (CMODE == 2) {
        acc = creg[i * 2 + j];
      } else {
#pragma unroll
        for (int r = 0; r < 8; ++r) acc[r] = 0.0f;
      }
#pragma unroll
      for (int kc = 0; kc < 8; ++kc)
        acc = __builtin_amdgcn_wmma_f32_16x16x4_f32(false, pf.f[i * 8 + kc], false,
                                                    bf.f[j * 8 + kc], (short)0, acc, false, false);
      *(v4f*)&S[yOff + cb]     = __builtin_shufflevector(acc, acc, 0,1,2,3);
      *(v4f*)&S[yOff + cb + 4] = __builtin_shufflevector(acc, acc, 4,5,6,7);
    }
  }
}

// ---------------- block-structured expm + propagation (one wave per batch element) ----------------
__global__ void __launch_bounds__(64) k_expm_prop(const float* __restrict__ Aws,
    const float* __restrict__ qdws, const float* __restrict__ z0,
    float* __restrict__ states, float* __restrict__ covs, int b0, int seg) {
  __shared__ float smem[2 * (7 * 1024 + 64)];
  int wave = threadIdx.x >> 5;
  int lane = threadIdx.x & 31;
  float* S    = smem + wave * (7 * 1024 + 64);
  float* zbuf = S + 7 * 1024;
  float* qs   = zbuf + 32;
  int bl = blockIdx.x * 2 + wave;      // chunk-local batch index
  int gb = b0 + bl;                    // global batch index
  int oAs = 0, oN = 1024, o11 = 2048, o12 = 3072, o22 = 4096, oT1 = 5120, oT2 = 6144;

  const float* Ab = Aws + (size_t)bl * 1024;
  // col-major(-As^T) == row-major(-As): coalesced global read
  for (int p = lane; p < 1024; p += 32) S[oN + p] = -Ab[p] * SQ_SCALE;
  qs[lane] = qdws[bl * 32 + lane] * SQ_SCALE;
  for (int p = lane; p < 1024; p += 32) {
    int r = p & 31, c = p >> 5;
    S[oAs + p] = -S[oN + r * 32 + c];                    // As col-major via LDS transpose
    float idv = (r == c) ? 1.0f : 0.0f;
    S[o11 + p] = idv; S[o12 + p] = 0.0f; S[o22 + p] = idv;
  }
  PF pfAs, pfN;
  loadPF(S, oAs, lane, pfAs);
  loadPF(S, oN, lane, pfN);
  // Horner: Rm = I + Ms@Rm/k with Ms=[[As,Qs],[0,-As^T]], Qs diagonal
  for (int k = 14; k >= 1; --k) {
    float rk = 1.0f / (float)k;
    BF bb;
    loadBF(S, o11, lane, bb);
    mm32x<0>(S, pfAs, bb, nullptr, -1, oT1, lane);
    loadBF(S, o12, lane, bb);
    mm32x<0>(S, pfAs, bb, nullptr, -1, oT2, lane);
    for (int p = lane; p < 1024; p += 32) {
      int r = p & 31, c = p >> 5;
      S[o11 + p] = ((r == c) ? 1.0f : 0.0f) + S[oT1 + p] * rk;
      S[o12 + p] = (S[oT2 + p] + qs[r] * S[o22 + p]) * rk;
    }
    loadBF(S, o22, lane, bb);
    mm32x<0>(S, pfN, bb, nullptr, -1, oT1, lane);
    for (int p = lane; p < 1024; p += 32) {
      int r = p & 31, c = p >> 5;
      S[o22 + p] = ((r == c) ? 1.0f : 0.0f) + S[oT1 + p] * rk;
    }
  }
  // 6 squarings of block-triangular Rm
  for (int sq = 0; sq < 6; ++sq) {
    PF pp; BF bb, bb2;
    loadPF(S, o11, lane, pp);
    loadBF(S, o11, lane, bb);
    mm32x<0>(S, pp, bb, nullptr, -1, oT1, lane);         // R11@R11
    loadBF(S, o12, lane, bb);
    mm32x<0>(S, pp, bb, nullptr, -1, oT2, lane);         // R11@R12
    loadPF(S, o12, lane, pp);
    loadBF(S, o22, lane, bb2);
    mm32x<1>(S, pp, bb2, nullptr, oT2, oT2, lane);       // += R12@R22
    loadPF(S, o22, lane, pp);
    mm32x<0>(S, pp, bb2, nullptr, -1, oAs, lane);        // R22@R22
    int t = o11; o11 = oT1; oT1 = t;
    t = o12; o12 = oT2; oT2 = t;
    t = o22; o22 = oAs; oAs = t;
  }
  // Qd = sym(Phi @ E12) -> o12 ; PhiT -> o22 (E22 dead)
  {
    PF pp; BF bb;
    loadPF(S, o11, lane, pp);
    loadBF(S, o12, lane, bb);
    mm32x<0>(S, pp, bb, nullptr, -1, oT1, lane);
    for (int p = lane; p < 1024; p += 32) {
      int r = p & 31, c = p >> 5;
      S[o12 + p] = 0.5f * (S[oT1 + p] + S[oT1 + r * 32 + c]);
    }
    for (int p = lane; p < 1024; p += 32) {
      int r = p & 31, c = p >> 5;
      S[o22 + p] = S[o11 + r * 32 + c];
    }
  }
  int oCov = oN, oU = oT1, oT = oT2;
  // cache loop-invariant operands for the 8 propagation steps
  PF pPhi;  loadPF(S, o11, lane, pPhi);
  BF bPhiT; loadBF(S, o22, lane, bPhiT);
  v8f cQd[4];
  {
    const int hi = lane >> 4, l15 = lane & 15;
#pragma unroll
    for (int i = 0; i < 2; ++i)
#pragma unroll
      for (int j = 0; j < 2; ++j) {
        int cb = (j * 16 + l15) * 32 + i * 16 + hi * 8;
        v4f c0 = *(const v4f*)&S[o12 + cb];
        v4f c1 = *(const v4f*)&S[o12 + cb + 4];
        cQd[i * 2 + j] = __builtin_shufflevector(c0, c1, 0,1,2,3,4,5,6,7);
      }
  }
  if (seg == 0) {
    zbuf[lane] = z0[(size_t)gb * 32 + lane];
    for (int p = lane; p < 1024; p += 32) S[oCov + p] = 0.0f;
  } else {
    int tp = seg * 8 - 1;
    zbuf[lane] = states[((size_t)gb * 24 + tp) * 32 + lane];
    for (int p = lane; p < 1024; p += 32)    // coalesced global read, transposed LDS store
      S[oCov + (p & 31) * 32 + (p >> 5)] = covs[((size_t)gb * 24 + tp) * 1024 + p];
  }
  for (int st = 0; st < 8; ++st) {
    float zi = 0.0f;
#pragma unroll
    for (int jj = 0; jj < 32; ++jj) zi += S[o11 + jj * 32 + lane] * zbuf[jj];  // conflict-free
    __builtin_amdgcn_wave_barrier();
    zbuf[lane] = zi;
    BF bCov; loadBF(S, oCov, lane, bCov);
    mm32x<0>(S, pPhi, bCov, nullptr, -1, oU, lane);      // U = Phi @ cov
    PF pU;   loadPF(S, oU, lane, pU);
    mm32x<2>(S, pU, bPhiT, cQd, -1, oT, lane);           // T = U @ PhiT + Qd
    for (int p = lane; p < 1024; p += 32) {
      int r = p & 31, c = p >> 5;
      S[oCov + p] = 0.5f * (S[oT + p] + S[oT + r * 32 + c]);
    }
    int tg = seg * 8 + st;
    states[((size_t)gb * 24 + tg) * 32 + lane] = zi;
    for (int p = lane; p < 1024; p += 32)    // coalesced global write
      covs[((size_t)gb * 24 + tg) * 1024 + p] = S[oCov + (p & 31) * 32 + (p >> 5)];
  }
}

// ---------------- host launcher ----------------
extern "C" void kernel_launch(void* const* d_in, const int* in_sizes, int n_in,
                              void* d_out, int out_size, void* d_ws, size_t ws_size,
                              hipStream_t stream) {
  (void)in_sizes; (void)n_in; (void)out_size; (void)ws_size;
  const float* context = (const float*)d_in[0];
  const float* z0    = (const float*)d_in[1];
  const float* S_W   = (const float*)d_in[2];
  const float* S_b   = (const float*)d_in[3];
  const float* L_W   = (const float*)d_in[4];
  const float* L_b   = (const float*)d_in[5];
  const float* G_W   = (const float*)d_in[6];
  const float* G_b   = (const float*)d_in[7];
  const float* Sig_W = (const float*)d_in[8];
  const float* Sig_b = (const float*)d_in[9];
  float* states = (float*)d_out;
  float* covsp  = states + (size_t)BB * 24 * 32;

  char* ws = (char*)d_ws;
  size_t off = 0;
  auto take = [&](size_t bytes) {
    off = (off + 255) & ~(size_t)255;
    void* p = ws + off;
    off += bytes;
    return p;
  };
  _Float16* ctx16 = (_Float16*)take((size_t)BB * CC * 2);
  _Float16* w16t  = (_Float16*)take((size_t)NPAD * CC * 2);
  float* Ybuf = (float*)take((size_t)BCH * NPAD * 4);
  float* Aws  = (float*)take((size_t)BCH * 1024 * 4);
  float* qdws = (float*)take((size_t)BCH * 32 * 4);

  k_cvt_ctx<<<(BB * CC + 255) / 256, 256, 0, stream>>>(context, ctx16);
  for (int s = 0; s < NSEG_; ++s) {
    k_build_w16t<<<(NPAD * CC + 255) / 256, 256, 0, stream>>>(S_W, L_W, G_W, Sig_W, w16t, s);
    for (int ch = 0; ch < NCHUNK; ++ch) {
      int b0 = ch * BCH;
      dim3 g1(BCH / 16, (NTILES + 3) / 4);
      k_gemm<<<g1, 128, 0, stream>>>(ctx16, w16t, Ybuf, b0);
      k_assemble<<<BCH, 128, 0, stream>>>(Ybuf, S_b, L_b, G_b, Sig_b, Aws, qdws, s);
      k_expm_prop<<<BCH / 2, 64, 0, stream>>>(Aws, qdws, z0, states, covsp, b0, s);
    }
  }
}
